// ILUVATARGroupGemmOp_22376779612784
// MI455X (gfx1250) — 8-bit matrix operand order repaired and checked by reading, compiled, not run on hardware
//
#include <hip/hip_runtime.h>
#include <cstdint>

typedef __attribute__((ext_vector_type(8))) int v8i;

#define G_      8
#define MG_     2048
#define K_      4096
#define N_      4096
#define BM_     256
#define BN_     128
#define BK_     64
#define LSTR_   80

__global__ __launch_bounds__(256) void group_gemm_i8_wmma(
    const int8_t* __restrict__ A,
    const int8_t* __restrict__ B,
    const float*  __restrict__ ls,
    const float*  __restrict__ rs,
    _Float16*     __restrict__ out)
{
    __shared__ __align__(16) signed char As[2][BM_ * LSTR_];
    __shared__ __align__(16) signed char Bs[2][BN_ * LSTR_];

    const int bn = blockIdx.x;
    const int bm = blockIdx.y;
    const int g  = blockIdx.z;

    const int tid   = threadIdx.x;
    const int wave  = tid >> 5;
    const int lane  = tid & 31;
    const int waveM = wave >> 1;
    const int waveN = wave & 1;
    const int l     = lane & 15;
    const int hi    = lane >> 4;

    const int8_t* Ag = A + (size_t)g * MG_ * K_ + (size_t)bm * BM_ * K_;
    const int8_t* Bg = B + (size_t)g * N_  * K_ + (size_t)bn * BN_ * K_;

    const int rr = tid >> 2;
    const int cc = (tid & 3) * 16;

    v8i acc[4][4] = {};

    auto gld = [&](const int8_t* base, int row, int k0) -> int4 {
        return *(const int4*)(base + (size_t)row * K_ + k0 + cc);
    };
    auto sst = [&](signed char* buf, int row, int4 v) {
        *(int4*)(buf + row * LSTR_ + cc) = v;
    };

    {
        int4 a0 = gld(Ag, rr, 0),       a1 = gld(Ag, rr + 64, 0);
        int4 a2 = gld(Ag, rr + 128, 0), a3 = gld(Ag, rr + 192, 0);
        int4 b0 = gld(Bg, rr, 0),       b1 = gld(Bg, rr + 64, 0);
        sst(As[0], rr, a0);       sst(As[0], rr + 64, a1);
        sst(As[0], rr + 128, a2); sst(As[0], rr + 192, a3);
        sst(Bs[0], rr, b0);       sst(Bs[0], rr + 64, b1);
    }
    __syncthreads();

    const int nK = K_ / BK_;
    #pragma unroll 1
    for (int kk = 0; kk < nK; ++kk) {
        const int cur  = kk & 1;
        const bool more = (kk + 1) < nK;

        int4 a0, a1, a2, a3, b0, b1;
        if (more) {
            const int k0 = (kk + 1) * BK_;
            a0 = gld(Ag, rr, k0);       a1 = gld(Ag, rr + 64, k0);
            a2 = gld(Ag, rr + 128, k0); a3 = gld(Ag, rr + 192, k0);
            b0 = gld(Bg, rr, k0);       b1 = gld(Bg, rr + 64, k0);
        }

        const signed char* Ab = As[cur];
        const signed char* Bb = Bs[cur];

        v8i bfrag[4];
        #pragma unroll
        for (int nt = 0; nt < 4; ++nt) {
            const signed char* brow = Bb + (waveN * 64 + nt * 16 + l) * LSTR_;
            union { v8i v; int4 q[2]; } bu;
            bu.q[0] = *(const int4*)(brow + hi * 16);
            bu.q[1] = *(const int4*)(brow + 32 + hi * 16);
            bfrag[nt] = bu.v;
        }

        #pragma unroll
        for (int mt = 0; mt < 4; ++mt) {
            const signed char* arow = Ab + (waveM * 64 + mt * 16 + l) * LSTR_;
            union { v8i v; long long d[4]; } au;
            au.d[0] = *(const long long*)(arow +  0 + hi * 16);
            au.d[1] = *(const long long*)(arow +  8 + hi * 16);
            au.d[2] = *(const long long*)(arow + 32 + hi * 16);
            au.d[3] = *(const long long*)(arow + 40 + hi * 16);
            #pragma unroll
            for (int nt = 0; nt < 4; ++nt) {
                acc[mt][nt] = __builtin_amdgcn_wmma_i32_16x16x64_iu8(
                     true, au.v,  true, bfrag[nt],
                    acc[mt][nt],  false,  false);
            }
        }

        if (more) {
            const int nxt = cur ^ 1;
            sst(As[nxt], rr, a0);       sst(As[nxt], rr + 64, a1);
            sst(As[nxt], rr + 128, a2); sst(As[nxt], rr + 192, a3);
            sst(Bs[nxt], rr, b0);       sst(Bs[nxt], rr + 64, b1);
        }
        __syncthreads();
    }

    const int colBase = bn * BN_ + waveN * 64;
    const int rowBase = g * MG_ + bm * BM_ + waveM * 64;
    #pragma unroll
    for (int nt = 0; nt < 4; ++nt) {
        const int col = colBase + nt * 16 + l;
        const float rsv = rs[(size_t)g * N_ + col];
        #pragma unroll
        for (int mt = 0; mt < 4; ++mt) {
            const int row0 = rowBase + mt * 16 + hi * 8;
            #pragma unroll
            for (int j = 0; j < 8; ++j) {
                const int row = row0 + j;
                const float v = (float)acc[mt][nt][j] * ls[row] * rsv;
                out[(size_t)row * N_ + col] = (_Float16)v;
            }
        }
    }
}

extern "C" void kernel_launch(void* const* d_in, const int* in_sizes, int n_in,
                              void* d_out, int out_size, void* d_ws, size_t ws_size,
                              hipStream_t stream) {
    const int8_t* A  = (const int8_t*)d_in[0];
    const int8_t* B  = (const int8_t*)d_in[1];
    const float*  ls = (const float*) d_in[2];
    const float*  rs = (const float*) d_in[3];
    _Float16* out = (_Float16*)d_out;

    dim3 grid(N_ / BN_, MG_ / BM_, G_);
    dim3 block(256, 1, 1);
    group_gemm_i8_wmma<<<grid, block, 0, stream>>>(A, B, ls, rs, out);
}
